// NeighborAware_20426864459837
// MI455X (gfx1250) — compile-verified, matrix operand order checked by reading, not run on hardware
//
#include <hip/hip_runtime.h>
#include <hip/hip_bf16.h>

typedef __attribute__((ext_vector_type(16))) __bf16 v16bf;
typedef __attribute__((ext_vector_type(8)))  float  v8f;
typedef __attribute__((ext_vector_type(4)))  float  fx4;   // native vector, no ctor

#define DD 128
#define KNB 5
#define TP 16           // pairs per block
#define NROWS (TP*KNB)  // 80 neighbor rows per tower

// packed-weight offsets in d_ws (bf16 elements). Each matrix is permuted into
// WMMA B-fragment order: [nt][ks][lane][16 bf16] (32 bytes per lane-fragment).
#define OFF_UWQ 0
#define OFF_UWK 16384
#define OFF_UWV 32768
#define OFF_UWO 49152
#define OFF_IWQ 65536
#define OFF_IWK 81920
#define OFF_IWV 98304
#define OFF_IWO 114688
#define OFF_W1  131072   // 8 nt x 8 ks
#define OFF_W2  163840   // 4 nt x 4 ks
#define PACK_ELEMS 172032

// ---- global-address-space load helpers (force global_load_*, not flat) -----
__device__ __forceinline__ fx4 gload4(const float* p) {
  return *(const __attribute__((address_space(1))) fx4*)p;
}
__device__ __forceinline__ float gloadf(const float* p) {
  return *(const __attribute__((address_space(1))) float*)p;
}
__device__ __forceinline__ v16bf gload_frag(const __bf16* p) {
  return *(const __attribute__((address_space(1))) v16bf*)p;   // 32B -> 2x b128
}

// A-matrix 16-bit layout (ISA 7.12.2): lane m (0..15) holds row M=m, K=0..7 in
// elems 0..7 and K=16..23 in elems 8..15; lane m+16 holds K=8..15 / 24..31.
// p points at row_base + ks*32 + kh*8. Loads batched (4x b128) then converted.
__device__ __forceinline__ v16bf frag_a_global(const float* p) {
  fx4 x0 = gload4(p), x1 = gload4(p + 4), y0 = gload4(p + 16), y1 = gload4(p + 20);
  v16bf f;
#pragma unroll
  for (int j = 0; j < 4; ++j) {
    f[j]      = (__bf16)x0[j];
    f[4 + j]  = (__bf16)x1[j];
    f[8 + j]  = (__bf16)y0[j];
    f[12 + j] = (__bf16)y1[j];
  }
  return f;
}

__device__ __forceinline__ v16bf frag_from_lds(const __bf16* p) {
  v16bf f;
#pragma unroll
  for (int j = 0; j < 8; ++j) f[j] = p[j];
#pragma unroll
  for (int j = 0; j < 8; ++j) f[8 + j] = p[16 + j];
  return f;
}

// packed B fragment (nt, ks) for a matrix with KS k-steps
__device__ __forceinline__ v16bf load_packed(const __bf16* base, int nt, int ks,
                                             int KS, int lane) {
  return gload_frag(base + ((size_t)((nt * KS + ks) * 32 + lane)) * 16);
}

// ---- [16 x 128] = gather(emb, rowids) @ Wpacked + bias -> LDS bf16 ----------
__device__ void mm16_gather(const float* __restrict__ emb,
                            const int* rowids,                 // 16 ids in LDS
                            const __bf16* __restrict__ Wp,     // packed fragments
                            const float* __restrict__ bias,
                            __bf16* out, int outLd, int lane) {
  const int mr = lane & 15, kh = lane >> 4;
  const int arow = rowids[mr];
  const float* rowp = emb + (size_t)arow * DD + kh * 8;
  v16bf a[4];
#pragma unroll
  for (int ks = 0; ks < 4; ++ks) a[ks] = frag_a_global(rowp + ks * 32);
#pragma unroll
  for (int nt = 0; nt < 8; ++nt) {
    v16bf b[4];
#pragma unroll
    for (int ks = 0; ks < 4; ++ks) b[ks] = load_packed(Wp, nt, ks, 4, lane);
    v8f acc = {};
#pragma unroll
    for (int ks = 0; ks < 4; ++ks)
      acc = __builtin_amdgcn_wmma_f32_16x16x32_bf16(false, a[ks], false, b[ks],
                                                    (short)0, acc, false, false);
    const float bv = gloadf(bias + nt * 16 + mr);
#pragma unroll
    for (int r = 0; r < 8; ++r)
      out[(size_t)(r + 8 * kh) * outLd + nt * 16 + mr] = (__bf16)(acc[r] + bv);
  }
}

// ---- [16 x ntCount*16] = A_lds[16 x KS*32] @ Wpacked + bias (opt. relu) -----
template <int KS>
__device__ void mm16_lds(const __bf16* A, int aLd,
                         const __bf16* __restrict__ Wp,
                         const float* __restrict__ bias,
                         __bf16* out, int outLd, int colOff,
                         int ntBase, int ntCount, bool relu, int lane) {
  const int mr = lane & 15, kh = lane >> 4;
  v16bf a[KS];
#pragma unroll
  for (int ks = 0; ks < KS; ++ks)
    a[ks] = frag_from_lds(A + (size_t)mr * aLd + ks * 32 + kh * 8);
  for (int nt = ntBase; nt < ntBase + ntCount; ++nt) {
    v16bf b[KS];
#pragma unroll
    for (int ks = 0; ks < KS; ++ks) b[ks] = load_packed(Wp, nt, ks, KS, lane);
    v8f acc = {};
#pragma unroll
    for (int ks = 0; ks < KS; ++ks)
      acc = __builtin_amdgcn_wmma_f32_16x16x32_bf16(false, a[ks], false, b[ks],
                                                    (short)0, acc, false, false);
    const float bv = gloadf(bias + nt * 16 + mr);
#pragma unroll
    for (int r = 0; r < 8; ++r) {
      float v = acc[r] + bv;
      if (relu) v = fmaxf(v, 0.f);
      out[(size_t)(r + 8 * kh) * outLd + nt * 16 + mr + colOff] = (__bf16)v;
    }
  }
}

// ---- weight pre-pack: fp32 row-major -> bf16 WMMA B-fragment layout ---------
__global__ __launch_bounds__(256)
void pack_weights(const float* __restrict__ src, __bf16* __restrict__ dst,
                  int ntiles, int ksteps, int ld) {
  const int idx = blockIdx.x * blockDim.x + threadIdx.x;       // one (frag,lane)
  const int total = ntiles * ksteps * 32;
  if (idx >= total) return;
  const int lane = idx & 31, frag = idx >> 5;
  const int ks = frag % ksteps, nt = frag / ksteps;
  const int mr = lane & 15, kh = lane >> 4;
  const float* p = src + (size_t)(nt * 16 + mr) * ld + ks * 32 + kh * 8;
  __bf16* q = dst + (size_t)idx * 16;
#pragma unroll
  for (int j = 0; j < 8; ++j) {
    q[j]     = (__bf16)gloadf(p + j);
    q[8 + j] = (__bf16)gloadf(p + 16 + j);
  }
}

// ---- fused NeighborAware forward -------------------------------------------
__global__ __launch_bounds__(256)
void neighbor_aware_fused(const int* __restrict__ user, const int* __restrict__ item,
                          const float* __restrict__ user_emb, const float* __restrict__ item_emb,
                          const int* __restrict__ user_topk, const int* __restrict__ item_topk,
                          const __bf16* __restrict__ wpack,
                          const float* __restrict__ ub, const float* __restrict__ ubo,
                          const float* __restrict__ ib, const float* __restrict__ ibo,
                          const float* __restrict__ b1, const float* __restrict__ b2,
                          const float* __restrict__ W3, const float* __restrict__ b3,
                          float* __restrict__ out, int B) {
  // ---- LDS carving (58.0 KB, heavy region reuse) ----
  __shared__ __align__(16) unsigned char smem[59392];
  __bf16* kvU = (__bf16*)(smem);              // [80][128] bf16 : K then V (tower U)
  __bf16* kvI = (__bf16*)(smem + 20480);      // [80][128] bf16 : K then V (tower I)
  __bf16* qU  = (__bf16*)(smem + 40960);      // [16][128] bf16
  __bf16* qI  = (__bf16*)(smem + 45056);      // [16][128] bf16
  __bf16* xb  = (__bf16*)(smem + 40960);      // [16][256] bf16 (aliases q, after scores)
  __bf16* ctU = (__bf16*)(smem + 49152);      // [16][128] bf16
  __bf16* ctI = (__bf16*)(smem + 53248);      // [16][128] bf16
  __bf16* h1  = (__bf16*)(smem + 49152);      // [16][128] bf16 (aliases ctx, after Wo)
  __bf16* h2  = (__bf16*)(smem + 53248);      // [16][64]  bf16
  float*  sc  = (float*)(smem + 57344);       // [2][16][8] scores -> softmax weights
  int*    trw = (int*)(smem + 58368);         // [2][16] target emb rows
  int*    nrw = (int*)(smem + 58496);         // [2][80] neighbor emb rows

  const int tid  = threadIdx.x;
  const int lane = tid & 31;
  const int wave = tid >> 5;
  const int base = blockIdx.x * TP;

  // S0: stage target and neighbor row ids
  if (tid < TP) {
    int p = min(base + tid, B - 1);
    trw[tid]      = user[p];
    trw[16 + tid] = item[p];
  }
  if (tid < NROWS) {
    int p = min(base + tid / KNB, B - 1);
    int n = tid % KNB;
    nrw[tid]      = user_topk[(size_t)user[p] * KNB + n];
    nrw[80 + tid] = item_topk[(size_t)item[p] * KNB + n];
  }
  __syncthreads();

  // S1: K-projection of all neighbor rows (10 jobs) + Q-projection (2 jobs)
  for (int job = wave; job < 12; job += 8) {
    if (job < 10) {
      const int tw = job / 5, t5 = job % 5;
      mm16_gather(tw ? item_emb : user_emb, nrw + tw * 80 + t5 * 16,
                  wpack + (tw ? OFF_IWK : OFF_UWK),
                  (tw ? ib : ub) + DD,
                  (tw ? kvI : kvU) + t5 * 16 * DD, DD, lane);
    } else {
      const int tw = job - 10;
      mm16_gather(tw ? item_emb : user_emb, trw + tw * 16,
                  wpack + (tw ? OFF_IWQ : OFF_UWQ), tw ? ib : ub,
                  tw ? qI : qU, DD, lane);
    }
  }
  __syncthreads();

  // S2: masked scaled-dot scores + softmax over K=5
  if (tid < 160) {
    const int tw = tid / 80, j = tid % 80, p = j / KNB, n = j % KNB;
    const __bf16* qp = (tw ? qI : qU) + p * DD;
    const __bf16* kp = (tw ? kvI : kvU) + j * DD;
    float s = 0.f;
#pragma unroll 8
    for (int d = 0; d < DD; ++d) s += (float)qp[d] * (float)kp[d];
    s *= 0.08838834764831845f;                 // 1/sqrt(128)
    if (nrw[tw * 80 + j] == 0) s = -1e30f;
    sc[tw * 128 + p * 8 + n] = s;
  }
  __syncthreads();
  if (tid < 32) {
    const int tw = tid / 16, p = tid % 16;
    float* s = sc + tw * 128 + p * 8;
    float mx = s[0];
#pragma unroll
    for (int n = 1; n < KNB; ++n) mx = fmaxf(mx, s[n]);
    float e[KNB], sum = 0.f;
#pragma unroll
    for (int n = 0; n < KNB; ++n) { e[n] = __expf(s[n] - mx); sum += e[n]; }
    const float inv = 1.f / sum;
#pragma unroll
    for (int n = 0; n < KNB; ++n) s[n] = e[n] * inv;
  }
  __syncthreads();

  // S3: V-projection overwrites the K buffers (10 jobs)
  for (int job = wave; job < 10; job += 8) {
    const int tw = job / 5, t5 = job % 5;
    mm16_gather(tw ? item_emb : user_emb, nrw + tw * 80 + t5 * 16,
                wpack + (tw ? OFF_IWV : OFF_UWV),
                (tw ? ib : ub) + 2 * DD,
                (tw ? kvI : kvU) + t5 * 16 * DD, DD, lane);
  }
  __syncthreads();

  // S4: ctx = softmax-weighted sum of V
  for (int e = tid; e < 2 * TP * DD; e += 256) {
    const int tw = e >> 11, rem = e & 2047, p = rem >> 7, d = rem & 127;
    const float* w = sc + tw * 128 + p * 8;
    const __bf16* v = (tw ? kvI : kvU) + p * KNB * DD + d;
    float acc = 0.f;
#pragma unroll
    for (int n = 0; n < KNB; ++n) acc += w[n] * (float)v[n * DD];
    (tw ? ctI : ctU)[p * DD + d] = (__bf16)acc;
  }
  __syncthreads();

  // S5: output projection Wo -> concatenated x[16][256] (u half | i half)
  if (wave < 2) {
    const int tw = wave;
    mm16_lds<4>(tw ? ctI : ctU, DD, wpack + (tw ? OFF_IWO : OFF_UWO),
                tw ? ibo : ubo, xb, 256, tw * 128, 0, 8, false, lane);
  }
  __syncthreads();

  // S6: h1 = relu(x @ W1^T + b1), one n-tile per wave (K = 256 -> 8 k-steps)
  mm16_lds<8>(xb, 256, wpack + OFF_W1, b1, h1, DD, 0, wave, 1, true, lane);
  __syncthreads();

  // S7: h2 = relu(h1 @ W2^T + b2), 4 n-tiles over waves 0..3
  if (wave < 4)
    mm16_lds<4>(h1, DD, wpack + OFF_W2, b2, h2, 64, 0, wave, 1, true, lane);
  __syncthreads();

  // S8: scalar head: out = h2 . W3 + b3
  if (tid < TP && base + tid < B) {
    const __bf16* hp = h2 + tid * 64;
    float s = gloadf(b3);
#pragma unroll 8
    for (int d = 0; d < 64; ++d) s += (float)hp[d] * gloadf(W3 + d);
    out[base + tid] = s;
  }
}

extern "C" void kernel_launch(void* const* d_in, const int* in_sizes, int n_in,
                              void* d_out, int out_size, void* d_ws, size_t ws_size,
                              hipStream_t stream) {
  const int*   user      = (const int*)  d_in[0];
  const int*   item      = (const int*)  d_in[1];
  const float* user_emb  = (const float*)d_in[2];
  const float* item_emb  = (const float*)d_in[3];
  const int*   user_topk = (const int*)  d_in[4];
  const int*   item_topk = (const int*)  d_in[5];
  const float* uW  = (const float*)d_in[6];
  const float* ub  = (const float*)d_in[7];
  const float* uWo = (const float*)d_in[8];
  const float* ubo = (const float*)d_in[9];
  const float* iW  = (const float*)d_in[10];
  const float* ib  = (const float*)d_in[11];
  const float* iWo = (const float*)d_in[12];
  const float* ibo = (const float*)d_in[13];
  const float* W1  = (const float*)d_in[14];
  const float* b1  = (const float*)d_in[15];
  const float* W2  = (const float*)d_in[16];
  const float* b2  = (const float*)d_in[17];
  const float* W3  = (const float*)d_in[18];
  const float* b3  = (const float*)d_in[19];

  __bf16* wpack = (__bf16*)d_ws;   // needs 344 KB; d_ws is plenty

  // pre-pack all weight matrices into WMMA B-fragment order (bf16)
  struct { const float* src; size_t off; int nt, ks, ld; } descs[10] = {
      {uW,               OFF_UWQ, 8, 4, DD},
      {uW + DD * DD,     OFF_UWK, 8, 4, DD},
      {uW + 2 * DD * DD, OFF_UWV, 8, 4, DD},
      {uWo,              OFF_UWO, 8, 4, DD},
      {iW,               OFF_IWQ, 8, 4, DD},
      {iW + DD * DD,     OFF_IWK, 8, 4, DD},
      {iW + 2 * DD * DD, OFF_IWV, 8, 4, DD},
      {iWo,              OFF_IWO, 8, 4, DD},
      {W1,               OFF_W1,  8, 8, 256},
      {W2,               OFF_W2,  4, 4, DD},
  };
  for (int i = 0; i < 10; ++i) {
    const int total = descs[i].nt * descs[i].ks * 32;
    pack_weights<<<(total + 255) / 256, 256, 0, stream>>>(
        descs[i].src, wpack + descs[i].off, descs[i].nt, descs[i].ks, descs[i].ld);
  }

  const int B = in_sizes[0];
  const int grid = (B + TP - 1) / TP;
  neighbor_aware_fused<<<grid, 256, 0, stream>>>(
      user, item, user_emb, item_emb, user_topk, item_topk,
      wpack, ub, ubo, ib, ibo, b1, b2, W3, b3,
      (float*)d_out, B);
}
